// HybridNet_90598040141938
// MI455X (gfx1250) — compile-verified, matrix operand order checked by reading, not run on hardware
//
#include <hip/hip_runtime.h>
#include <hip/hip_bf16.h>

typedef _Float16 half8 __attribute__((ext_vector_type(8)));
typedef _Float16 v16h  __attribute__((ext_vector_type(16)));
typedef float    v8f   __attribute__((ext_vector_type(8)));

#define B_   64
#define N0_  1024
#define EPG_ 16384
#define C_   256

// ---------------------------------------------------------------- utils
static __device__ __forceinline__ float warp_red(float v) {
  v += __shfl_xor(v, 16);
  v += __shfl_xor(v, 8);
  v += __shfl_xor(v, 4);
  v += __shfl_xor(v, 2);
  v += __shfl_xor(v, 1);
  return v;
}

static __device__ __forceinline__ v16h ldfrag(const _Float16* __restrict__ p) {
  half8 lo = *(const half8*)p;
  half8 hi = *(const half8*)(p + 16);
  v16h f;
#pragma unroll
  for (int i = 0; i < 8; i++) { f[i] = lo[i]; f[8 + i] = hi[i]; }
  return f;
}

// ---------------------------------------------------------------- WMMA GEMM
// D[M,N] = act( sum_k A[m,k]*Bw[n,k] + bias[n] ), A/Bw row-major f16.
// One wave computes a 32x32 output tile (2x2 WMMA register tiling).
// Out-of-range rows are CLAMPED (not masked): row m of D depends only on
// row m of A / row n of Bw, and out-of-range results are never stored, so
// clamping keeps loads legal, EXEC all-ones, and the inner loop branch-free.
__global__ void __launch_bounds__(32)
wmma_gemm_rr(const _Float16* __restrict__ A, int lda, long strideA,
             const _Float16* __restrict__ Bw, int ldb,
             float* __restrict__ D, int ldd, long strideD,
             const float* __restrict__ bias,
             int M, int N, int K, int relu)
{
  const int tm = blockIdx.x, tn = blockIdx.y, bz = blockIdx.z;
  A += (size_t)bz * strideA;
  D += (size_t)bz * strideD;
  const int lane = threadIdx.x;
  const int r = lane & 15, g = lane >> 4;

  int ar0 = tm * 32 + r;        if (ar0 > M - 1) ar0 = M - 1;
  int ar1 = tm * 32 + 16 + r;   if (ar1 > M - 1) ar1 = M - 1;
  int br0 = tn * 32 + r;        if (br0 > N - 1) br0 = N - 1;
  int br1 = tn * 32 + 16 + r;   if (br1 > N - 1) br1 = N - 1;

  const _Float16* Ap0 = A + (size_t)ar0 * lda + 8 * g;
  const _Float16* Ap1 = A + (size_t)ar1 * lda + 8 * g;
  const _Float16* Bp0 = Bw + (size_t)br0 * ldb + 8 * g;
  const _Float16* Bp1 = Bw + (size_t)br1 * ldb + 8 * g;

  v8f c00 = {}, c01 = {}, c10 = {}, c11 = {};
  for (int k0 = 0; k0 < K; k0 += 32) {
    v16h a0 = ldfrag(Ap0 + k0);
    v16h a1 = ldfrag(Ap1 + k0);
    v16h b0 = ldfrag(Bp0 + k0);
    v16h b1 = ldfrag(Bp1 + k0);
    __builtin_prefetch(Ap0 + k0 + 128, 0, 1);
    __builtin_prefetch(Ap1 + k0 + 128, 0, 1);
    c00 = __builtin_amdgcn_wmma_f32_16x16x32_f16(false, a0, false, b0, (short)0, c00, false, false);
    c01 = __builtin_amdgcn_wmma_f32_16x16x32_f16(false, a0, false, b1, (short)0, c01, false, false);
    c10 = __builtin_amdgcn_wmma_f32_16x16x32_f16(false, a1, false, b0, (short)0, c10, false, false);
    c11 = __builtin_amdgcn_wmma_f32_16x16x32_f16(false, a1, false, b1, (short)0, c11, false, false);
  }

  // store: tile (mt, nt); lane (r,g) vgpr j holds D[mt + j + 8g][nt + r]
#pragma unroll
  for (int tj = 0; tj < 4; tj++) {
    const v8f& c = (tj == 0) ? c00 : (tj == 1) ? c01 : (tj == 2) ? c10 : c11;
    const int mt = tm * 32 + ((tj >> 1) ? 16 : 0);
    const int nt = tn * 32 + ((tj & 1) ? 16 : 0);
    const int nn = nt + r;
    if (nn < N) {
      const float bv = bias[nn];
#pragma unroll
      for (int j = 0; j < 8; j++) {
        const int m = mt + j + 8 * g;
        if (m < M) {
          float v = c[j] + bv;
          if (relu) v = fmaxf(v, 0.0f);
          D[(size_t)m * ldd + nn] = v;
        }
      }
    }
  }
}

// ---------------------------------------------------------------- conversions
__global__ void cvt_f16_flat(const float* __restrict__ s, _Float16* __restrict__ d, long n) {
  long i = (long)blockIdx.x * 256 + threadIdx.x;
  if (i < n) d[i] = (_Float16)s[i];
}

__global__ void cvt_f16_pad(const float* __restrict__ s, _Float16* __restrict__ d,
                            int rows, int Ks, int Kd) {
  long i = (long)blockIdx.x * 256 + threadIdx.x;
  if (i >= (long)rows * Kd) return;
  int rr = (int)(i / Kd), c = (int)(i % Kd);
  d[i] = (c < Ks) ? (_Float16)s[(long)rr * Ks + c] : (_Float16)0.0f;
}

// convert per-graph n rows (B_ graphs, stride N0_*C_)
__global__ void cvt_x_f16(const float* __restrict__ s, _Float16* __restrict__ d, int n) {
  long i = (long)blockIdx.x * 256 + threadIdx.x;
  long tot = (long)B_ * n * C_;
  if (i >= tot) return;
  long b = i / ((long)n * C_);
  long rem = i % ((long)n * C_);
  long off = b * (long)N0_ * C_ + rem;
  d[off] = (_Float16)s[off];
}

__global__ void relu_x_kernel(float* X, int n) {
  long i = (long)blockIdx.x * 256 + threadIdx.x;
  long tot = (long)B_ * n * C_;
  if (i >= tot) return;
  long b = i / ((long)n * C_);
  long rem = i % ((long)n * C_);
  float* p = X + b * (long)N0_ * C_ + rem;
  *p = fmaxf(*p, 0.0f);
}

// ---------------------------------------------------------------- graph kernels
__global__ void edge_init_kernel(const int* __restrict__ ei, int* row, int* col, int* em) {
  int e = blockIdx.x * 256 + threadIdx.x;
  if (e >= B_ * EPG_) return;
  int b = e >> 14;
  row[e] = ei[e] - b * N0_;
  col[e] = ei[B_ * EPG_ + e] - b * N0_;
  em[e] = 1;
}

__global__ void deg_kernel(const int* __restrict__ col, const int* __restrict__ em,
                           float* deg) {
  int e = blockIdx.x * 256 + threadIdx.x;
  if (e >= B_ * EPG_) return;
  if (!em[e]) return;
  int b = e >> 14;
  atomicAdd(&deg[b * N0_ + col[e]], 1.0f);
}

__global__ void dinv_kernel(const float* __restrict__ deg, float* dinv) {
  int i = blockIdx.x * 256 + threadIdx.x;
  if (i >= B_ * N0_) return;
  float d = deg[i];
  dinv[i] = (d > 0.0f) ? rsqrtf(d) : 0.0f;
}

// agg[b,c,:] += dinv[r]*dinv[c] * h[b,r,:]  (64 lanes x 4ch per edge)
__global__ void scatter_kernel(const float* __restrict__ Hb,
                               const int* __restrict__ row, const int* __restrict__ col,
                               const int* __restrict__ em, const float* __restrict__ dinv,
                               float* agg) {
  long tid = (long)blockIdx.x * 256 + threadIdx.x;
  long tot = (long)B_ * EPG_ * 64;
  if (tid >= tot) return;
  int e = (int)(tid >> 6);
  int lc = (int)(tid & 63);
  if (!em[e]) return;
  int b = e >> 14;
  int r = row[e], c = col[e];
  float nrm = dinv[b * N0_ + r] * dinv[b * N0_ + c];
  const float4 hv = *(const float4*)(Hb + ((size_t)b * N0_ + r) * C_ + lc * 4);
  float* dst = agg + ((size_t)b * N0_ + c) * C_ + lc * 4;
  atomicAdd(dst + 0, nrm * hv.x);
  atomicAdd(dst + 1, nrm * hv.y);
  atomicAdd(dst + 2, nrm * hv.z);
  atomicAdd(dst + 3, nrm * hv.w);
}

__global__ void wnorm_kernel(const float* __restrict__ w, float* invn) {
  int lane = threadIdx.x;
  float s = 0.0f;
  for (int i = 0; i < 8; i++) { float v = w[lane + 32 * i]; s += v * v; }
  s = warp_red(s);
  if (lane == 0) invn[0] = rsqrtf(s);
}

__global__ void score_kernel(const float* __restrict__ X, const float* __restrict__ w,
                             const float* __restrict__ invn, float* score, int n) {
  int wid = (blockIdx.x * blockDim.x + threadIdx.x) >> 5;
  int lane = threadIdx.x & 31;
  if (wid >= B_ * n) return;
  int b = wid / n, m = wid % n;
  const float* p = X + ((size_t)b * N0_ + m) * C_;
  float s = 0.0f;
#pragma unroll
  for (int i = 0; i < 8; i++) { int c = lane * 8 + i; s += p[c] * w[c]; }
  s = warp_red(s);
  if (lane == 0) score[b * N0_ + m] = tanhf(s * invn[0]);
}

// bitonic top-k: one block (1024 thr) per graph; descending, ties -> lower index
__global__ void __launch_bounds__(1024)
topk_sort(const float* __restrict__ score, int n, int k,
          int* perm, float* vals, int* newid) {
  __shared__ float ss[1024];
  __shared__ int ii[1024];
  int b = blockIdx.x, t = threadIdx.x;
  ss[t] = (t < n) ? score[b * N0_ + t] : -INFINITY;
  ii[t] = t;
  __syncthreads();
  for (int kk = 2; kk <= 1024; kk <<= 1) {
    for (int j = kk >> 1; j > 0; j >>= 1) {
      int x = t ^ j;
      if (x > t) {
        float s1 = ss[t], s2 = ss[x];
        int i1 = ii[t], i2 = ii[x];
        bool desc = ((t & kk) == 0);
        bool firstSmaller = (s1 < s2) || (s1 == s2 && i1 > i2);
        bool doswap = desc ? firstSmaller : !firstSmaller;
        if (doswap) { ss[t] = s2; ss[x] = s1; ii[t] = i2; ii[x] = i1; }
      }
      __syncthreads();
    }
  }
  newid[b * N0_ + t] = -1;
  __syncthreads();
  if (t < k) {
    perm[b * N0_ + t] = ii[t];
    vals[b * N0_ + t] = ss[t];
    newid[b * N0_ + ii[t]] = t;
  }
}

__global__ void pool_gather(const float* __restrict__ Xin, float* __restrict__ Xout,
                            const int* __restrict__ perm, const float* __restrict__ vals) {
  int b = blockIdx.y, m = blockIdx.x, c = threadIdx.x;
  int src = perm[b * N0_ + m];
  float v = vals[b * N0_ + m];
  Xout[((size_t)b * N0_ + m) * C_ + c] = Xin[((size_t)b * N0_ + src) * C_ + c] * v;
}

__global__ void remap_kernel(int* row, int* col, int* em, const int* __restrict__ newid) {
  int e = blockIdx.x * 256 + threadIdx.x;
  if (e >= B_ * EPG_) return;
  int b = e >> 14;
  int r2 = newid[b * N0_ + row[e]];
  int c2 = newid[b * N0_ + col[e]];
  int v = em[e] && (r2 >= 0) && (c2 >= 0);
  row[e] = r2 > 0 ? r2 : 0;
  col[e] = c2 > 0 ? c2 : 0;
  em[e] = v;
}

__global__ void feats_kernel(const float* __restrict__ X, float* sx, int k) {
  int i = blockIdx.x * 256 + threadIdx.x;
  if (i >= B_ * C_) return;
  int b = i >> 8, c = i & 255;
  const float* p = X + ((size_t)b * N0_) * C_ + c;
  float mx = -INFINITY, sm = 0.0f;
  for (int m = 0; m < k; m++) {
    float v = p[(size_t)m * C_];
    mx = fmaxf(mx, v);
    sm += v;
  }
  atomicAdd(&sx[b * 512 + c], 0.2f * mx);
  atomicAdd(&sx[b * 512 + 256 + c], 0.2f * sm / (float)k);
}

// ---------------------------------------------------------------- attention MLP + xf
__global__ void __launch_bounds__(512)
attmlp_kernel(const float* __restrict__ sx, float* __restrict__ zcat,
              const float* W0, const float* B0, const float* W1, const float* B1,
              const float* W2, const float* B2, const float* W3, const float* B3,
              const float* W4, const float* B4, const float* W5, const float* B5) {
  __shared__ float a0[512], a1[512];
  int b = blockIdx.x, t = threadIdx.x;
  a0[t] = sx[b * 512 + t];
  __syncthreads();
  if (t < 64) { float s = B0[t]; for (int i = 0; i < 512; i++) s += a0[i] * W0[t * 512 + i]; a1[t] = fmaxf(s, 0.f); }
  __syncthreads();
  if (t < 16) { float s = B1[t]; for (int i = 0; i < 64; i++) s += a1[i] * W1[t * 64 + i]; a0[t] = fmaxf(s, 0.f); }
  __syncthreads();
  if (t < 2)  { float s = B2[t]; for (int i = 0; i < 16; i++) s += a0[i] * W2[t * 16 + i]; a1[t] = fmaxf(s, 0.f); }
  __syncthreads();
  if (t < 16) { float s = B3[t]; for (int i = 0; i < 2;  i++) s += a1[i] * W3[t * 2 + i];  a0[t] = fmaxf(s, 0.f); }
  __syncthreads();
  if (t < 64) { float s = B4[t]; for (int i = 0; i < 16; i++) s += a0[i] * W4[t * 16 + i]; a1[t] = fmaxf(s, 0.f); }
  __syncthreads();
  {
    float s = B5[t];
    for (int i = 0; i < 64; i++) s += a1[i] * W5[t * 64 + i];
    float a = fmaxf(s, 0.f);
    zcat[b * 1024 + t] = (a + 1.0f) * sx[b * 512 + t];
  }
}

// ---------------------------------------------------------------- conv-branch post ops
__global__ void chanstat_kernel(const float* __restrict__ U, float* umax, float* umean, int Hf) {
  long i = (long)blockIdx.x * 256 + threadIdx.x;
  if (i >= (long)B_ * Hf) return;
  const float* p = U + i * 64;
  float mx = -INFINITY, sm = 0.0f;
#pragma unroll 8
  for (int c = 0; c < 64; c++) { float v = p[c]; mx = fmaxf(mx, v); sm += v; }
  umax[i] = mx;
  umean[i] = sm * (1.0f / 64.0f);
}

__global__ void attout_kernel(const float* __restrict__ umax, const float* __restrict__ umean,
                              const float* __restrict__ K2, const float* __restrict__ b2,
                              float* __restrict__ tc, int Hf, int fs, int lo, int coloff) {
  long i = (long)blockIdx.x * 256 + threadIdx.x;
  if (i >= (long)B_ * Hf) return;
  int b = (int)(i / Hf), y = (int)(i % Hf);
  const float* mx = umax + (size_t)b * Hf;
  const float* mn = umean + (size_t)b * Hf;
  float s = b2[0];
  for (int dy = 0; dy < fs; dy++) {
    int yy = y - lo + dy;
    if (yy >= 0 && yy < Hf) s += mx[yy] * K2[dy] + mn[yy] * K2[fs + dy];
  }
  float att = 1.0f / (1.0f + expf(-s));
  tc[(size_t)b * 3936 + coloff + y] = att * mx[y];
}

// ---------------------------------------------------------------- head
__global__ void head_kernel(const float* __restrict__ Z, const float* __restrict__ W,
                            const float* __restrict__ bb, float* out) {
  int b = threadIdx.x;
  if (b >= B_) return;
  float z[6];
  float m = -INFINITY;
  for (int j = 0; j < 6; j++) {
    float s = bb[j];
    for (int i = 0; i < 64; i++) s += Z[b * 64 + i] * W[j * 64 + i];
    z[j] = s;
    m = fmaxf(m, s);
  }
  float lse = 0.0f;
  for (int j = 0; j < 6; j++) lse += expf(z[j] - m);
  lse = m + logf(lse);
  for (int j = 0; j < 6; j++) out[b * 6 + j] = z[j] - lse;
}

// ---------------------------------------------------------------- host
extern "C" void kernel_launch(void* const* d_in, const int* in_sizes, int n_in,
                              void* d_out, int out_size, void* d_ws, size_t ws_size,
                              hipStream_t stream) {
  (void)in_sizes; (void)n_in; (void)out_size; (void)ws_size;

  const float* x_in = (const float*)d_in[0];
  const float* t_in = (const float*)d_in[1];
  const float* gcnW[5]; const float* gcnB[5]; const float* poolW[5];
  for (int i = 0; i < 5; i++) {
    gcnW[i] = (const float*)d_in[2 + i];
    gcnB[i] = (const float*)d_in[7 + i];
    poolW[i] = (const float*)d_in[12 + i];
  }
  const float* attW[6]; const float* attB[6];
  for (int i = 0; i < 6; i++) { attW[i] = (const float*)d_in[17 + i]; attB[i] = (const float*)d_in[23 + i]; }
  const float* convK[8]; const float* convB[8]; const float* a2K[8]; const float* a2B[8];
  for (int i = 0; i < 8; i++) {
    convK[i] = (const float*)d_in[29 + i]; convB[i] = (const float*)d_in[37 + i];
    a2K[i] = (const float*)d_in[45 + i];   a2B[i] = (const float*)d_in[53 + i];
  }
  const float* linW[5]; const float* linB[5];
  for (int i = 0; i < 5; i++) { linW[i] = (const float*)d_in[61 + i]; linB[i] = (const float*)d_in[66 + i]; }
  const int* edge_index = (const int*)d_in[71];
  float* out = (float*)d_out;

  // ---- workspace carve-up
  char* p = (char*)d_ws;
  auto alloc = [&](size_t bytes) -> void* {
    void* r = (void*)p;
    p += (bytes + 255) & ~(size_t)255;
    return r;
  };
  const size_t szX = (size_t)B_ * N0_ * C_ * 4;
  float* X0 = (float*)alloc(szX);
  float* X1 = (float*)alloc(szX);
  float* HB = (float*)alloc(szX);
  _Float16* XH = (_Float16*)alloc(szX / 2);
  _Float16* WH = (_Float16*)alloc((size_t)512 * 3936 * 2);
  _Float16* TH = (_Float16*)alloc((size_t)B_ * 500 * 128 * 2);
  float* U = (float*)alloc((size_t)B_ * 3930 * 64 * 4);
  float* UMX = (float*)alloc((size_t)B_ * 3930 * 4);
  float* UMN = (float*)alloc((size_t)B_ * 3930 * 4);
  float* TC = (float*)alloc((size_t)B_ * 3936 * 4);
  _Float16* ZH = (_Float16*)alloc((size_t)B_ * 3936 * 2);
  float* DEG = (float*)alloc((size_t)B_ * N0_ * 4);
  float* DINV = (float*)alloc((size_t)B_ * N0_ * 4);
  float* SCORE = (float*)alloc((size_t)B_ * N0_ * 4);
  float* VALS = (float*)alloc((size_t)B_ * N0_ * 4);
  int* PERM = (int*)alloc((size_t)B_ * N0_ * 4);
  int* NEWID = (int*)alloc((size_t)B_ * N0_ * 4);
  int* ROW = (int*)alloc((size_t)B_ * EPG_ * 4);
  int* COL = (int*)alloc((size_t)B_ * EPG_ * 4);
  int* EM = (int*)alloc((size_t)B_ * EPG_ * 4);
  float* SX = (float*)alloc((size_t)B_ * 512 * 4);
  float* ZC = (float*)alloc((size_t)B_ * 1024 * 4);
  float* Z1 = (float*)alloc((size_t)B_ * 512 * 4);
  float* Z2 = (float*)alloc((size_t)B_ * 128 * 4);
  float* Z3 = (float*)alloc((size_t)B_ * 64 * 4);
  float* INVN = (float*)alloc(256);

  auto blocks = [](long n) { return (unsigned)((n + 255) / 256); };
  auto gemm = [&](const _Float16* A, int lda, long sA, const _Float16* Bw, int ldb,
                  float* D, int ldd, long sD, const float* bias,
                  int M, int N, int K, int relu, int batch) {
    dim3 g((M + 31) / 32, (N + 31) / 32, batch);
    wmma_gemm_rr<<<g, 32, 0, stream>>>(A, lda, sA, Bw, ldb, D, ldd, sD, bias, M, N, K, relu);
  };

  // =========================== GNN branch ===========================
  hipMemcpyAsync(X0, x_in, szX, hipMemcpyDeviceToDevice, stream);
  edge_init_kernel<<<blocks(B_ * EPG_), 256, 0, stream>>>(edge_index, ROW, COL, EM);
  hipMemsetAsync(SX, 0, (size_t)B_ * 512 * 4, stream);

  const int nseq[5] = {1024, 820, 656, 525, 420};
  const int kseq[5] = {820, 656, 525, 420, 336};

  for (int L = 0; L < 5; L++) {
    const int n = nseq[L], k = kseq[L];
    // h = x @ W^T + b
    cvt_f16_flat<<<blocks((long)C_ * C_), 256, 0, stream>>>(gcnW[L], WH, (long)C_ * C_);
    cvt_x_f16<<<blocks((long)B_ * n * C_), 256, 0, stream>>>(X0, XH, n);
    gemm(XH, C_, (long)N0_ * C_, WH, C_, HB, C_, (long)N0_ * C_, gcnB[L],
         n, C_, C_, /*relu=*/0, B_);
    // normalized aggregation into X1
    hipMemsetAsync(X1, 0, szX, stream);
    hipMemsetAsync(DEG, 0, (size_t)B_ * N0_ * 4, stream);
    deg_kernel<<<blocks(B_ * EPG_), 256, 0, stream>>>(COL, EM, DEG);
    dinv_kernel<<<blocks(B_ * N0_), 256, 0, stream>>>(DEG, DINV);
    scatter_kernel<<<blocks((long)B_ * EPG_ * 64), 256, 0, stream>>>(HB, ROW, COL, EM, DINV, X1);
    relu_x_kernel<<<blocks((long)B_ * n * C_), 256, 0, stream>>>(X1, n);
    // top-k pool
    wnorm_kernel<<<1, 32, 0, stream>>>(poolW[L], INVN);
    score_kernel<<<blocks((long)B_ * n * 32), 256, 0, stream>>>(X1, poolW[L], INVN, SCORE, n);
    topk_sort<<<B_, 1024, 0, stream>>>(SCORE, n, k, PERM, VALS, NEWID);
    pool_gather<<<dim3(k, B_), 256, 0, stream>>>(X1, X0, PERM, VALS);
    remap_kernel<<<blocks(B_ * EPG_), 256, 0, stream>>>(ROW, COL, EM, NEWID);
    feats_kernel<<<blocks(B_ * C_), 256, 0, stream>>>(X0, SX, k);
  }

  // attention MLP -> xf into ZC[:, 0:512]
  attmlp_kernel<<<B_, 512, 0, stream>>>(SX, ZC,
      attW[0], attB[0], attW[1], attB[1], attW[2], attB[2],
      attW[3], attB[3], attW[4], attB[4], attW[5], attB[5]);

  // =========================== conv branch ===========================
  const int F[8]  = {2, 4, 6, 8, 10, 12, 16, 20};
  const int Hh[8] = {499, 497, 495, 493, 491, 489, 485, 481};
  const int FS[8] = {250, 249, 248, 247, 246, 245, 243, 241};
  int cumH[9]; cumH[0] = 0;
  for (int i = 0; i < 8; i++) cumH[i + 1] = cumH[i] + Hh[i];

  cvt_f16_flat<<<blocks((long)B_ * 500 * 128), 256, 0, stream>>>(t_in, TH, (long)B_ * 500 * 128);
  hipMemsetAsync(TC, 0, (size_t)B_ * 3936 * 4, stream);

  for (int i = 0; i < 8; i++) {
    const int K = F[i] * 128, Hf = Hh[i];
    float* Uf = U + (size_t)B_ * cumH[i] * 64;
    float* mxf = UMX + (size_t)B_ * cumH[i];
    float* mnf = UMN + (size_t)B_ * cumH[i];
    cvt_f16_flat<<<blocks((long)64 * K), 256, 0, stream>>>(convK[i], WH, (long)64 * K);
    // u = relu(conv): overlapping-window GEMM, A row stride = 128
    gemm(TH, 128, (long)500 * 128, WH, K, Uf, 64, (long)Hf * 64, convB[i],
         Hf, 64, K, /*relu=*/1, B_);
    chanstat_kernel<<<blocks((long)B_ * Hf), 256, 0, stream>>>(Uf, mxf, mnf, Hf);
    const int lo = (FS[i] - 1) / 2;
    attout_kernel<<<blocks((long)B_ * Hf), 256, 0, stream>>>(mxf, mnf, a2K[i], a2B[i],
                                                             TC, Hf, FS[i], lo, cumH[i]);
  }

  // tc = relu(tc @ W0^T + b0) -> ZC[:, 512:1024]   (K padded 3930 -> 3936)
  cvt_f16_flat<<<blocks((long)B_ * 3936), 256, 0, stream>>>(TC, ZH, (long)B_ * 3936);
  cvt_f16_pad<<<blocks((long)512 * 3936), 256, 0, stream>>>(linW[0], WH, 512, 3930, 3936);
  gemm(ZH, 3936, 0, WH, 3936, ZC + 512, 1024, 0, linB[0], B_, 512, 3936, 1, 1);

  // =========================== head MLP ===========================
  cvt_f16_flat<<<blocks((long)B_ * 1024), 256, 0, stream>>>(ZC, ZH, (long)B_ * 1024);
  cvt_f16_flat<<<blocks((long)512 * 1024), 256, 0, stream>>>(linW[1], WH, (long)512 * 1024);
  gemm(ZH, 1024, 0, WH, 1024, Z1, 512, 0, linB[1], B_, 512, 1024, 1, 1);

  cvt_f16_flat<<<blocks((long)B_ * 512), 256, 0, stream>>>(Z1, ZH, (long)B_ * 512);
  cvt_f16_flat<<<blocks((long)128 * 512), 256, 0, stream>>>(linW[2], WH, (long)128 * 512);
  gemm(ZH, 512, 0, WH, 512, Z2, 128, 0, linB[2], B_, 128, 512, 1, 1);

  cvt_f16_flat<<<blocks((long)B_ * 128), 256, 0, stream>>>(Z2, ZH, (long)B_ * 128);
  cvt_f16_flat<<<blocks((long)64 * 128), 256, 0, stream>>>(linW[3], WH, (long)64 * 128);
  gemm(ZH, 128, 0, WH, 128, Z3, 64, 0, linB[3], B_, 64, 128, 1, 1);

  head_kernel<<<1, 64, 0, stream>>>(Z3, linW[4], linB[4], out);
}
